// M3Site_7937099563210
// MI455X (gfx1250) — matrix operand order repaired and checked by reading, not compiled, not run on hardware
//
#include <hip/hip_runtime.h>
#include <hip/hip_bf16.h>

// ---------------- problem constants ----------------
#define G_    32
#define P_    1024
#define NN_   16384
#define E_    1280
#define D_    256
#define H_    256
#define C_    7
#define FF_   128
#define COND_ 768
#define GP_   (G_ * P_)   // 32768 padded rows

// ---------------- WMMA types ----------------
typedef __attribute__((ext_vector_type(16))) __bf16 v16bf;
typedef __attribute__((ext_vector_type(8)))  float  v8f;

__device__ __forceinline__ unsigned f2bf_bits(float f) {
    union { float f; unsigned u; } v; v.f = f;
    unsigned u = v.u;
    return (u + 0x7FFFu + ((u >> 16) & 1u)) >> 16;   // round-to-nearest-even
}
__device__ __forceinline__ uint2 pack4_bf(float a, float b, float c, float d) {
    uint2 r;
    r.x = f2bf_bits(a) | (f2bf_bits(b) << 16);
    r.y = f2bf_bits(c) | (f2bf_bits(d) << 16);
    return r;
}

// ============================================================
// WMMA GEMM:  C = act( alpha * A @ op(B) + bias )
//   B_IS_NT = true : B is [N,K] row-major (A @ B^T)  -> linear layers, Q@K^T
//   B_IS_NT = false: B is [K,N] row-major (A @ B)    -> probs @ V
//   NGUARD  = true : N not multiple of 64 (fc2, N=7) -> bounds checks
// 128x64x32 block tiles, 256 threads = 8 wave32 waves.
// Wave w: wm = w&3 (32-row strip), wn = w>>2 (32-col strip)
//   -> 2 A-frags x 2 B-frags -> 4 x v_wmma_f32_16x16x32_bf16 per K-step.
// Register double-buffering: tile k+1 global loads are issued while the
// WMMAs for tile k execute; f32->bf16 conversion happens at LDS-store time.
// ============================================================
template<bool B_IS_NT, bool NGUARD>
__global__ void wmma_gemm(const float* __restrict__ A, long long sA,
                          const float* __restrict__ Bm, long long sB,
                          const float* __restrict__ bias,
                          float* __restrict__ Cm, long long sC,
                          int M, int N, int K, float alpha, int act)
{
    __shared__ unsigned short As[128][40];   // [m][k], +8 pad
    __shared__ unsigned short Bs[64][40];    // [n][k], +8 pad

    const int b = blockIdx.z;
    const float* Ab = A  + (long long)b * sA;
    const float* Bb = Bm + (long long)b * sB;
    float*       Cb = Cm + (long long)b * sC;

    const int m0  = blockIdx.y * 128;        // M is always a multiple of 128 here
    const int n0  = blockIdx.x * 64;
    const int tid = threadIdx.x;
    const int wave = tid >> 5;
    const int lane = tid & 31;
    const int wm = wave & 3;                 // 32-row strip
    const int wn = wave >> 2;                // 32-col strip
    const int half = lane >> 4;
    const int mr = lane & 15;

    // per-thread staging coordinates
    const int ar_[1] = {0}; (void)ar_;
    const int aR = tid >> 3, aC4 = (tid & 7) * 4;          // A: 4 rows aR, aR+32, aR+64, aR+96
    const int bR = tid >> 3, bC4 = (tid & 7) * 4;          // B(NT): rows bR, bR+32
    const int cK = tid >> 4, cN4 = (tid & 15) * 4;         // B(NN): k rows cK, cK+16

    auto loadA = [&](int k0, float4* ar) {
        #pragma unroll
        for (int it = 0; it < 4; ++it)
            ar[it] = *(const float4*)(Ab + (long long)(m0 + aR + it * 32) * K + (k0 + aC4));
    };
    auto storeA = [&](const float4* ar) {
        #pragma unroll
        for (int it = 0; it < 4; ++it)
            *(uint2*)&As[aR + it * 32][aC4] = pack4_bf(ar[it].x, ar[it].y, ar[it].z, ar[it].w);
    };
    auto loadB = [&](int k0, float4* br) {
        if constexpr (B_IS_NT) {
            #pragma unroll
            for (int it = 0; it < 2; ++it) {
                br[it] = make_float4(0.f, 0.f, 0.f, 0.f);
                if (!NGUARD || (n0 + bR + it * 32 < N))
                    br[it] = *(const float4*)(Bb + (long long)(n0 + bR + it * 32) * K + (k0 + bC4));
            }
        } else {
            #pragma unroll
            for (int it = 0; it < 2; ++it)
                br[it] = *(const float4*)(Bb + (long long)(k0 + cK + it * 16) * N + (n0 + cN4));
        }
    };
    auto storeB = [&](const float4* br) {
        if constexpr (B_IS_NT) {
            #pragma unroll
            for (int it = 0; it < 2; ++it)
                *(uint2*)&Bs[bR + it * 32][bC4] = pack4_bf(br[it].x, br[it].y, br[it].z, br[it].w);
        } else {
            #pragma unroll
            for (int it = 0; it < 2; ++it) {
                int c = cK + it * 16;
                Bs[cN4 + 0][c] = (unsigned short)f2bf_bits(br[it].x);
                Bs[cN4 + 1][c] = (unsigned short)f2bf_bits(br[it].y);
                Bs[cN4 + 2][c] = (unsigned short)f2bf_bits(br[it].z);
                Bs[cN4 + 3][c] = (unsigned short)f2bf_bits(br[it].w);
            }
        }
    };

    v8f acc00 = {0.f,0.f,0.f,0.f,0.f,0.f,0.f,0.f};
    v8f acc01 = acc00, acc10 = acc00, acc11 = acc00;

    float4 aReg[4], bReg[2];
    loadA(0, aReg);
    loadB(0, bReg);

    for (int k0 = 0; k0 < K; k0 += 32) {
        // commit staged tile k0 to LDS (f32 -> bf16 at store time)
        storeA(aReg);
        storeB(bReg);
        __syncthreads();

        // issue global loads for tile k0+32 now; they complete under the
        // WMMAs below and are only waited on at the next storeA/storeB.
        if (k0 + 32 < K) {
            loadA(k0 + 32, aReg);
            loadB(k0 + 32, bReg);
        }
        // prefetch two tiles ahead into L2 (global_prefetch_b8)
        if (k0 + 64 < K)
            __builtin_prefetch(Ab + (long long)(m0 + (tid >> 1)) * K + (k0 + 64) + (tid & 1) * 16, 0, 1);

        // --- A fragments: 16-bit A 16x32 layout (ISA 7.12.2) ---
        v16bf af0, af1;
        #pragma unroll
        for (int v = 0; v < 8; ++v) {
            int kb = (v < 4 ? 2 * v : 16 + 2 * (v - 4)) + 8 * half;
            int r0 = wm * 32 + mr;
            int r1 = wm * 32 + 16 + mr;
            af0[2 * v]     = __builtin_bit_cast(__bf16, As[r0][kb]);
            af0[2 * v + 1] = __builtin_bit_cast(__bf16, As[r0][kb + 1]);
            af1[2 * v]     = __builtin_bit_cast(__bf16, As[r1][kb]);
            af1[2 * v + 1] = __builtin_bit_cast(__bf16, As[r1][kb + 1]);
        }
        // --- B fragments: lane = column, K striped ---
        v16bf bf0, bf1;
        #pragma unroll
        for (int v = 0; v < 8; ++v) {
            int kb = (v < 4 ? 2 * v : 16 + 2 * (v - 4)) + 8 * half;
            int c0 = wn * 32 + mr;
            int c1 = wn * 32 + 16 + mr;
            bf0[2 * v]     = __builtin_bit_cast(__bf16, Bs[c0][kb]);
            bf0[2 * v + 1] = __builtin_bit_cast(__bf16, Bs[c0][kb + 1]);
            bf1[2 * v]     = __builtin_bit_cast(__bf16, Bs[c1][kb]);
            bf1[2 * v + 1] = __builtin_bit_cast(__bf16, Bs[c1][kb + 1]);
        }
        acc00 = __builtin_amdgcn_wmma_f32_16x16x32_bf16(false, af0, false, bf0, (short)0, acc00, false, false);
        acc01 = __builtin_amdgcn_wmma_f32_16x16x32_bf16(false, af0, false, bf1, (short)0, acc01, false, false);
        acc10 = __builtin_amdgcn_wmma_f32_16x16x32_bf16(false, af1, false, bf0, (short)0, acc10, false, false);
        acc11 = __builtin_amdgcn_wmma_f32_16x16x32_bf16(false, af1, false, bf1, (short)0, acc11, false, false);
        __syncthreads();
    }

    // --- epilogue: C layout (VGPR r: lanes0-15 M=r, lanes16-31 M=r+8; N=lane&15) ---
    const int colA = n0 + wn * 32 + mr;
    const int colB = n0 + wn * 32 + 16 + mr;
    const float biasA = (!NGUARD || colA < N) ? (bias ? bias[colA] : 0.f) : 0.f;
    const float biasB = (!NGUARD || colB < N) ? (bias ? bias[colB] : 0.f) : 0.f;
    #pragma unroll
    for (int r = 0; r < 8; ++r) {
        int row0 = m0 + wm * 32 + r + (half ? 8 : 0);
        int row1 = row0 + 16;
        if (!NGUARD || colA < N) {
            float v0 = acc00[r] * alpha + biasA;
            float v1 = acc10[r] * alpha + biasA;
            if (act == 1) { v0 = fmaxf(v0, 0.f); v1 = fmaxf(v1, 0.f); }
            Cb[(long long)row0 * N + colA] = v0;
            Cb[(long long)row1 * N + colA] = v1;
        }
        if (!NGUARD || colB < N) {
            float v0 = acc01[r] * alpha + biasB;
            float v1 = acc11[r] * alpha + biasB;
            if (act == 1) { v0 = fmaxf(v0, 0.f); v1 = fmaxf(v1, 0.f); }
            Cb[(long long)row0 * N + colB] = v0;
            Cb[(long long)row1 * N + colB] = v1;
        }
    }
}

// ============================================================
// elementwise / reduction helper kernels
// ============================================================
__global__ void zero_f(float* p, long long n) {
    for (long long i = (long long)blockIdx.x * blockDim.x + threadIdx.x; i < n;
         i += (long long)gridDim.x * blockDim.x) p[i] = 0.f;
}

__global__ void count_kernel(const int* __restrict__ batch, int* counts) {
    int i = blockIdx.x * blockDim.x + threadIdx.x;
    if (i < NN_) atomicAdd(&counts[batch[i]], 1);
}

__global__ void scan_starts(const int* counts, int* starts) {
    if (threadIdx.x == 0 && blockIdx.x == 0) {
        int acc = 0;
        for (int g = 0; g < G_; ++g) { starts[g] = acc; acc += counts[g]; }
    }
}

__global__ void pos_kernel(const int* __restrict__ batch, const int* __restrict__ starts,
                           int* __restrict__ pos) {
    int i = blockIdx.x * blockDim.x + threadIdx.x;
    if (i < NN_) pos[i] = i - starts[batch[i]];
}

__global__ void scatter_kernel(const float* __restrict__ src, float* __restrict__ dst,
                               const int* __restrict__ batch, const int* __restrict__ pos, int W) {
    long long total = (long long)NN_ * W;
    for (long long idx = (long long)blockIdx.x * blockDim.x + threadIdx.x; idx < total;
         idx += (long long)gridDim.x * blockDim.x) {
        int n = (int)(idx / W);
        int w = (int)(idx % W);
        long long row = (long long)batch[n] * P_ + pos[n];
        dst[row * W + w] = src[idx];
    }
}

__global__ void softmax_rows(float* __restrict__ s, int cols) {
    __shared__ float red[256];
    float* p = s + (long long)blockIdx.x * cols;
    int tid = threadIdx.x;
    float mx = -3.402823466e38f;
    for (int c = tid; c < cols; c += 256) mx = fmaxf(mx, p[c]);
    red[tid] = mx; __syncthreads();
    for (int o = 128; o > 0; o >>= 1) { if (tid < o) red[tid] = fmaxf(red[tid], red[tid + o]); __syncthreads(); }
    mx = red[0]; __syncthreads();
    float sum = 0.f;
    for (int c = tid; c < cols; c += 256) { float e = __expf(p[c] - mx); p[c] = e; sum += e; }
    red[tid] = sum; __syncthreads();
    for (int o = 128; o > 0; o >>= 1) { if (tid < o) red[tid] += red[tid + o]; __syncthreads(); }
    float inv = 1.f / red[0];
    for (int c = tid; c < cols; c += 256) p[c] *= inv;
}

__global__ void add_ln(const float* __restrict__ a, const float* __restrict__ b,
                       const float* __restrict__ g, const float* __restrict__ be,
                       float* __restrict__ o, int dim) {
    __shared__ float red[256];
    long long row = blockIdx.x;
    const float* pa = a + row * dim;
    const float* pb = b + row * dim;
    float* po = o + row * dim;
    int tid = threadIdx.x;
    float s = 0.f;
    for (int c = tid; c < dim; c += 256) s += pa[c] + pb[c];
    red[tid] = s; __syncthreads();
    for (int o2 = 128; o2 > 0; o2 >>= 1) { if (tid < o2) red[tid] += red[tid + o2]; __syncthreads(); }
    float mean = red[0] / dim; __syncthreads();
    float v = 0.f;
    for (int c = tid; c < dim; c += 256) { float d = pa[c] + pb[c] - mean; v += d * d; }
    red[tid] = v; __syncthreads();
    for (int o2 = 128; o2 > 0; o2 >>= 1) { if (tid < o2) red[tid] += red[tid + o2]; __syncthreads(); }
    float inv = rsqrtf(red[0] / dim + 1e-5f);
    for (int c = tid; c < dim; c += 256)
        po[c] = (pa[c] + pb[c] - mean) * inv * g[c] + be[c];
}

__global__ void concat_cond(const float* __restrict__ q, const float* __restrict__ func,
                            float* __restrict__ cat, int dim, int catW) {
    long long total = (long long)GP_ * catW;
    for (long long idx = (long long)blockIdx.x * blockDim.x + threadIdx.x; idx < total;
         idx += (long long)gridDim.x * blockDim.x) {
        long long row = idx / catW;
        int c = (int)(idx % catW);
        cat[idx] = (c < dim) ? q[row * dim + c]
                             : func[(row / P_) * COND_ + (c - dim)];
    }
}

// wfc dot-product (N=1) + sigmoid + gated fusion, writes out[row, 0:1536]
__global__ void gate_out(const float* __restrict__ sss, const float* __restrict__ s2,
                         const float* __restrict__ esm, const float* __restrict__ egnn,
                         const float* __restrict__ wfcw, const float* __restrict__ wfcb,
                         float* __restrict__ outp) {
    __shared__ float red[256];
    long long row = blockIdx.x;
    int tid = threadIdx.x;
    const float* ps = sss  + row * (long long)D_;
    const float* p2 = s2   + row * (long long)E_;
    const float* pe = esm  + row * (long long)E_;
    const float* pg = egnn + row * (long long)D_;
    float acc = 0.f;
    for (int k = tid; k < 2 * (E_ + D_); k += 256) {
        float v;
        if      (k < 256)  v = ps[k];
        else if (k < 1536) v = p2[k - 256];
        else if (k < 2816) v = pe[k - 1536];
        else               v = pg[k - 2816];
        acc += wfcw[k] * v;
    }
    red[tid] = acc; __syncthreads();
    for (int o = 128; o > 0; o >>= 1) { if (tid < o) red[tid] += red[tid + o]; __syncthreads(); }
    float wgt = 1.f / (1.f + __expf(-(red[0] + wfcb[0])));
    float* po = outp + row * (long long)(E_ + D_);
    for (int k = tid; k < (E_ + D_); k += 256) {
        float fu = (k < 256)  ? ps[k] : p2[k - 256];
        float to = (k < 1280) ? pe[k] : pg[k - 1280];
        po[k] = wgt * fu + (1.f - wgt) * to;
    }
}

__global__ void bn_stats(const float* __restrict__ h, float* bsum, float* bsq) {
    int c = threadIdx.x;                       // channel 0..255
    long long base = (long long)blockIdx.x * 128 * H_;
    float s = 0.f, q = 0.f;
    for (int r = 0; r < 128; ++r) {
        float v = h[base + (long long)r * H_ + c];
        s += v; q += v * v;
    }
    atomicAdd(&bsum[c], s);
    atomicAdd(&bsq[c], q);
}

__global__ void bn_finalize(const float* bsum, const float* bsq, float* mu, float* var) {
    int c = threadIdx.x;
    if (c < H_) {
        float m = bsum[c] / (float)GP_;
        mu[c] = m;
        var[c] = bsq[c] / (float)GP_ - m * m;   // biased, matches jnp.var
    }
}

__global__ void bn_apply_relu(float* __restrict__ h, const float* mu, const float* var,
                              const float* g, const float* b) {
    long long total = (long long)GP_ * H_;
    for (long long idx = (long long)blockIdx.x * blockDim.x + threadIdx.x; idx < total;
         idx += (long long)gridDim.x * blockDim.x) {
        int c = (int)(idx % H_);
        float v = (h[idx] - mu[c]) * rsqrtf(var[c] + 1e-5f) * g[c] + b[c];
        h[idx] = fmaxf(v, 0.f);
    }
}

// gather logits -> recon; center-loss accumulate; token softmax -> d_out
__global__ void token_cl(const float* __restrict__ logits, const int* __restrict__ batch,
                         const int* __restrict__ pos, const int* __restrict__ y,
                         const float* __restrict__ centers, float* __restrict__ token_out,
                         float* clacc) {
    int i = blockIdx.x * blockDim.x + threadIdx.x;
    if (i >= NN_) return;
    long long r = (long long)(batch[i] * P_ + pos[i]) * C_;
    const float* ct = centers + y[i] * C_;
    float v[C_];
    float mx = -3.402823466e38f, cl = 0.f;
    #pragma unroll
    for (int c = 0; c < C_; ++c) {
        v[c] = logits[r + c];
        mx = fmaxf(mx, v[c]);
        float d = v[c] - ct[c];
        cl += d * d;
    }
    atomicAdd(clacc, cl);
    float s = 0.f;
    #pragma unroll
    for (int c = 0; c < C_; ++c) { v[c] = __expf(v[c] - mx); s += v[c]; }
    float inv = 1.f / s;
    #pragma unroll
    for (int c = 0; c < C_; ++c) token_out[(long long)i * C_ + c] = v[c] * inv;
}

__global__ void finalize_losses(const float* __restrict__ centers, const float* clacc,
                                float* out_cl, float* out_il) {
    __shared__ float red[32];
    int t = threadIdx.x;
    float h = 0.f;
    const int NP = C_ * (C_ - 1) / 2;   // 21
    if (t < NP) {
        int c = t, i = 0, j = 1;
        for (i = 0; i < C_; ++i) {
            int cnt = C_ - 1 - i;
            if (c < cnt) { j = i + 1 + c; break; }
            c -= cnt;
        }
        float s = 0.f;
        for (int k = 0; k < C_; ++k) {
            float d = centers[i * C_ + k] - centers[j * C_ + k];
            s += d * d;
        }
        float dist = sqrtf(s + 1e-12f);
        h = fmaxf(0.1f - dist, 0.f);
    }
    red[t] = h; __syncthreads();
    for (int o = 16; o > 0; o >>= 1) { if (t < o) red[t] += red[t + o]; __syncthreads(); }
    if (t == 0) {
        *out_il = red[0] / (float)NP;
        *out_cl = clacc[0] / (float)NN_;
    }
}

// ============================================================
// host-side orchestration
// ============================================================
extern "C" void kernel_launch(void* const* d_in, const int* in_sizes, int n_in,
                              void* d_out, int out_size, void* d_ws, size_t ws_size,
                              hipStream_t stream) {
    (void)in_sizes; (void)n_in; (void)out_size; (void)ws_size;

    const float* esm_rep = (const float*)d_in[0];
    const float* egnn_out= (const float*)d_in[1];
    const float* func    = (const float*)d_in[2];
    const int*   batch   = (const int*)d_in[3];
    const int*   y       = (const int*)d_in[4];
    const float* fc1_w   = (const float*)d_in[5];
    const float* fc1_b   = (const float*)d_in[6];
    const float* bn_g    = (const float*)d_in[7];
    const float* bn_b    = (const float*)d_in[8];
    const float* fc2_w   = (const float*)d_in[9];
    const float* fc2_b   = (const float*)d_in[10];
    const float* f1_kw   = (const float*)d_in[11];
    const float* f1_kb   = (const float*)d_in[12];
    const float* f1_vw   = (const float*)d_in[13];
    const float* f1_vb   = (const float*)d_in[14];
    const float* f1_ow   = (const float*)d_in[15];
    const float* f1_ob   = (const float*)d_in[16];
    const float* f1_l1g  = (const float*)d_in[17];
    const float* f1_l1b  = (const float*)d_in[18];
    const float* f1_f1w  = (const float*)d_in[19];
    const float* f1_f1b  = (const float*)d_in[20];
    const float* f1_f2w  = (const float*)d_in[21];
    const float* f1_f2b  = (const float*)d_in[22];
    const float* f1_l2g  = (const float*)d_in[23];
    const float* f1_l2b  = (const float*)d_in[24];
    const float* f2_kw   = (const float*)d_in[25];
    const float* f2_kb   = (const float*)d_in[26];
    const float* f2_vw   = (const float*)d_in[27];
    const float* f2_vb   = (const float*)d_in[28];
    const float* f2_ow   = (const float*)d_in[29];
    const float* f2_ob   = (const float*)d_in[30];
    const float* f2_l1g  = (const float*)d_in[31];
    const float* f2_l1b  = (const float*)d_in[32];
    const float* f2_f1w  = (const float*)d_in[33];
    const float* f2_f1b  = (const float*)d_in[34];
    const float* f2_f2w  = (const float*)d_in[35];
    const float* f2_f2b  = (const float*)d_in[36];
    const float* f2_l2g  = (const float*)d_in[37];
    const float* f2_l2b  = (const float*)d_in[38];
    const float* wfc_w   = (const float*)d_in[39];
    const float* wfc_b   = (const float*)d_in[40];
    const float* centers = (const float*)d_in[41];

    float* W = (float*)d_ws;

    // ---- workspace layout (floats), with time-disjoint aliasing ----
    const long long GP = GP_;
    long long o_esm  = 0;                           // [GP,1280]
    long long o_egnn = o_esm  + GP * E_;            // [GP,256]
    long long o_t0   = o_egnn + GP * D_;            // [GP,1280] temp
    long long o_t1   = o_t0   + GP * E_;            // [GP,1280] temp
    long long o_t2   = o_t1   + GP * E_;            // [GP,1280] temp
    long long o_sss  = o_t2   + GP * E_;            // [GP,256]
    long long o_s2   = o_sss  + GP * D_;            // [GP,1280]
    long long o_cat  = o_s2   + GP * E_;            // [GP,2048]; scores alias here
    long long o_ffh  = o_cat  + GP * 2048;          // [GP,128]
    long long o_misc = o_ffh  + GP * FF_;

    float* esm_pad  = W + o_esm;
    float* egnn_pad = W + o_egnn;
    float* t0       = W + o_t0;
    float* t1       = W + o_t1;
    float* t2       = W + o_t2;
    float* sssb     = W + o_sss;
    float* s2b      = W + o_s2;
    float* catb     = W + o_cat;
    float* scores   = W + o_cat;                    // alias (dead before cat is live)
    float* ffh      = W + o_ffh;
    float* outb     = W + o_t0;                     // [GP,1536] spans t0..t1 (dead then)
    float* hbuf     = W + o_t2;                     // [GP,256]
    float* logits   = W + o_t2 + GP * D_;           // [GP,7]

    int*   counts = (int*)(W + o_misc);             // 32
    int*   starts = counts + 32;                    // 32
    float* bnsum  = W + o_misc + 64;                // 256
    float* bnsq   = bnsum + 256;                    // 256
    float* bnmu   = bnsq  + 256;                    // 256
    float* bnvar  = bnmu  + 256;                    // 256
    float* clacc  = bnvar + 256;                    // 1
    int*   pos    = (int*)(W + o_misc + 2048);      // 16384

    float* tok_out = (float*)d_out;                 // [N,7] ++ cl ++ il

    const dim3 blk(256);
    auto gemmNT = [&](const float* A, long long sA, const float* B, long long sB,
                      const float* bias, float* Cp, long long sC,
                      int M, int Nn, int K, float alpha, int act, int bat) {
        dim3 grid(Nn / 64, M / 128, bat);
        wmma_gemm<true, false><<<grid, blk, 0, stream>>>(A, sA, B, sB, bias, Cp, sC, M, Nn, K, alpha, act);
    };
    auto gemmNN = [&](const float* A, long long sA, const float* B, long long sB,
                      const float* bias, float* Cp, long long sC,
                      int M, int Nn, int K, float alpha, int act, int bat) {
        dim3 grid(Nn / 64, M / 128, bat);
        wmma_gemm<false, false><<<grid, blk, 0, stream>>>(A, sA, B, sB, bias, Cp, sC, M, Nn, K, alpha, act);
    };

    // ---- Stage A: zero + ragged->padded scatter ----
    zero_f<<<4096, blk, 0, stream>>>(esm_pad,  GP * (long long)E_);
    zero_f<<<1024, blk, 0, stream>>>(egnn_pad, GP * (long long)D_);
    zero_f<<<8,    blk, 0, stream>>>(W + o_misc, 1152);   // counts/starts/bn/clacc
    count_kernel<<<NN_ / 256, blk, 0, stream>>>(batch, counts);
    scan_starts<<<1, 32, 0, stream>>>(counts, starts);
    pos_kernel<<<NN_ / 256, blk, 0, stream>>>(batch, starts, pos);
    scatter_kernel<<<4096, blk, 0, stream>>>(esm_rep,  esm_pad,  batch, pos, E_);
    scatter_kernel<<<1024, blk, 0, stream>>>(egnn_out, egnn_pad, batch, pos, D_);

    // ---- Stage B: sss = funicross(Q=egnn, K=V=esm, dim=D) ----
    gemmNT(esm_pad, 0, f1_kw, 0, f1_kb, t0, 0, GP_, D_, E_, 1.f, 0, 1);          // Kp1
    gemmNT(esm_pad, 0, f1_vw, 0, f1_vb, t1, 0, GP_, D_, E_, 1.f, 0, 1);          // Vp1
    gemmNT(egnn_pad, (long long)P_ * D_, t0, (long long)P_ * D_, nullptr,
           scores, (long long)P_ * P_, P_, P_, D_, 1.f / sqrtf((float)D_), 0, G_);
    softmax_rows<<<GP_, blk, 0, stream>>>(scores, P_);
    gemmNN(scores, (long long)P_ * P_, t1, (long long)P_ * D_, nullptr,
           t2, (long long)P_ * D_, P_, D_, P_, 1.f, 0, G_);                       // ctx
    gemmNT(t2, 0, f1_ow, 0, f1_ob, t0, 0, GP_, D_, D_, 1.f, 0, 1);               // out-proj
    add_ln<<<GP_, blk, 0, stream>>>(egnn_pad, t0, f1_l1g, f1_l1b, t1, D_);       // Qn1
    concat_cond<<<4096, blk, 0, stream>>>(t1, func, catb, D_, D_ + COND_);
    gemmNT(catb, 0, f1_f1w, 0, f1_f1b, ffh, 0, GP_, FF_, D_ + COND_, 1.f, 1, 1); // relu
    gemmNT(ffh, 0, f1_f2w, 0, f1_f2b, t0, 0, GP_, D_, FF_, 1.f, 0, 1);
    add_ln<<<GP_, blk, 0, stream>>>(t1, t0, f1_l2g, f1_l2b, sssb, D_);           // sss

    // ---- Stage C: s2 = funicross(Q=esm, K=V=egnn, dim=E) ----
    gemmNT(egnn_pad, 0, f2_kw, 0, f2_kb, t0, 0, GP_, E_, D_, 1.f, 0, 1);         // Kp2
    gemmNT(egnn_pad, 0, f2_vw, 0, f2_vb, t1, 0, GP_, E_, D_, 1.f, 0, 1);         // Vp2
    gemmNT(esm_pad, (long long)P_ * E_, t0, (long long)P_ * E_, nullptr,
           scores, (long long)P_ * P_, P_, P_, E_, 1.f / sqrtf((float)E_), 0, G_);
    softmax_rows<<<GP_, blk, 0, stream>>>(scores, P_);
    gemmNN(scores, (long long)P_ * P_, t1, (long long)P_ * E_, nullptr,
           t2, (long long)P_ * E_, P_, E_, P_, 1.f, 0, G_);                       // ctx
    gemmNT(t2, 0, f2_ow, 0, f2_ob, t0, 0, GP_, E_, E_, 1.f, 0, 1);               // out-proj
    add_ln<<<GP_, blk, 0, stream>>>(esm_pad, t0, f2_l1g, f2_l1b, t1, E_);        // Qn2
    concat_cond<<<4096, blk, 0, stream>>>(t1, func, catb, E_, E_ + COND_);
    gemmNT(catb, 0, f2_f1w, 0, f2_f1b, ffh, 0, GP_, FF_, E_ + COND_, 1.f, 1, 1); // relu
    gemmNT(ffh, 0, f2_f2w, 0, f2_f2b, t0, 0, GP_, E_, FF_, 1.f, 0, 1);
    add_ln<<<GP_, blk, 0, stream>>>(t1, t0, f2_l2g, f2_l2b, s2b, E_);            // s2

    // ---- Stage D: gated fusion -> fc1 -> BN -> relu -> fc2 ----
    gate_out<<<GP_, blk, 0, stream>>>(sssb, s2b, esm_pad, egnn_pad, wfc_w, wfc_b, outb);
    gemmNT(outb, 0, fc1_w, 0, fc1_b, hbuf, 0, GP_, H_, E_ + D_, 1.f, 0, 1);      // fc1
    bn_stats<<<GP_ / 128, blk, 0, stream>>>(hbuf, bnsum, bnsq);
    bn_finalize<<<1, H_, 0, stream>>>(bnsum, bnsq, bnmu, bnvar);
    bn_apply_relu<<<2048, blk, 0, stream>>>(hbuf, bnmu, bnvar, bn_g, bn_b);
    {   // fc2: N=7 -> guarded variant
        dim3 grid(1, GP_ / 128, 1);
        wmma_gemm<true, true><<<grid, blk, 0, stream>>>(hbuf, 0, fc2_w, 0, fc2_b,
                                                        logits, 0, GP_, C_, H_, 1.f, 0);
    }

    // ---- Stage E: gather + losses + token softmax ----
    token_cl<<<NN_ / 256, blk, 0, stream>>>(logits, batch, pos, y, centers, tok_out, clacc);
    finalize_losses<<<1, 32, 0, stream>>>(centers, clacc,
                                          tok_out + (long long)NN_ * C_,
                                          tok_out + (long long)NN_ * C_ + 1);
}
